// SelfModifyingRecurrent_33921651704677
// MI455X (gfx1250) — matrix operand order repaired and checked by reading, not compiled, not run on hardware
//
#include <hip/hip_runtime.h>
#include <hip/hip_bf16.h>
#include <math.h>

// ---------------------------------------------------------------------------
// SelfModifyingRecurrent for MI455X (gfx1250), wave32 + WMMA f16 (f32 accum).
//
// Algebraic simplifications (exact):
//   * h_prev == 0 for all layers  ->  gh = b_hh[l], h = (1-z)*n. W_hh unused.
//   * rank-1 weight modification  ->  per-row scalar s[b] = cond *
//         (MOD_SCALE/B) * dot(cur[b], colsum(sig)) added to every gi column.
//
// v2: all WMMA operands pre-converted to f16 (weights once per call, cur once
// per layer) so GEMM inner loops are pure b128-load + v_wmma — no f32->f16
// cvt VALU traffic and half the L2 bytes vs v1.
// ---------------------------------------------------------------------------

#define B_DIM 8192
#define E_DIM 1024
#define H_DIM 512
#define L_NUM 3
#define MOD_SCALE 0.01f

typedef __attribute__((ext_vector_type(16))) _Float16 v16h;
typedef __attribute__((ext_vector_type(8)))  float    v8f;

// ---------------- WMMA fragment helpers (16x16x32, f16 in / f32 out) -------
// A (16x32, MxK) per ISA 7.12.2: lane m=lane&15, g=lane>>4:
//   h[0..7]  = A[m][k0 + 8g .. +7],  h[8..15] = A[m][k0 + 16 + 8g .. +7]
__device__ __forceinline__ v16h frag_a_f16(const _Float16* __restrict__ A, int lda,
                                           int row0, int k0, int lane) {
  const int m = lane & 15, g = lane >> 4;
  const _Float16* p = A + (size_t)(row0 + m) * lda + k0 + g * 8;
  v16h h;
#pragma unroll
  for (int i = 0; i < 8; ++i) h[i] = p[i];
#pragma unroll
  for (int i = 0; i < 8; ++i) h[8 + i] = p[16 + i];
  return h;
}

// B (32x16, KxN) for out = cur @ W.T with W stored (N,K) row-major:
//   B[k][n] = W[n][k]; lane n=lane&15, g=lane>>4 needs W[n0+n][k0+8g .. +7]
//   and W[n0+n][k0+16+8g .. +7] (two b128 loads, no transpose needed).
__device__ __forceinline__ v16h frag_b_wT_f16(const _Float16* __restrict__ W, int ldw,
                                              int n0, int k0, int lane) {
  const int n = lane & 15, g = lane >> 4;
  const _Float16* p = W + (size_t)(n0 + n) * ldw + k0 + g * 8;
  v16h h;
#pragma unroll
  for (int i = 0; i < 8; ++i) h[i] = p[i];
#pragma unroll
  for (int i = 0; i < 8; ++i) h[8 + i] = p[16 + i];
  return h;
}

__device__ __forceinline__ v8f wmma16(v16h a, v16h b, v8f c) {
  return __builtin_amdgcn_wmma_f32_16x16x32_f16(false, a, false, b, (short)0, c,
                                                false, false);
}

__device__ __forceinline__ float sigmoidf_(float x) {
  return 1.0f / (1.0f + expf(-x));
}

// ---------------- conversion kernel (f32 -> f16, 8 elems/thread) -----------
__global__ __launch_bounds__(256) void cvt_f16_kernel(const float* __restrict__ src,
                                                      _Float16* __restrict__ dst,
                                                      unsigned long long n) {
  unsigned long long i =
      ((unsigned long long)blockIdx.x * blockDim.x + threadIdx.x) * 8ull;
  if (i < n) {
#pragma unroll
    for (int j = 0; j < 8; ++j) dst[i + j] = (_Float16)src[i + j];
  }
}

// ---------------- small kernels --------------------------------------------
__global__ void zero_kernel(float* __restrict__ colsum, float* __restrict__ gatesum) {
  int i = blockIdx.x * blockDim.x + threadIdx.x;
  if (i < E_DIM) colsum[i] = 0.0f;
  if (i == E_DIM) gatesum[0] = 0.0f;
}

// sum_b sigmoid(dot(cur[b], Wg) + bg)  ->  *gatesum (atomic)
__global__ __launch_bounds__(256) void gate_kernel(const float* __restrict__ cur,
                                                   const float* __restrict__ Wg,
                                                   const float* __restrict__ bg,
                                                   float* __restrict__ gatesum) {
  const int lane = threadIdx.x & 31;
  const int gw = (blockIdx.x * blockDim.x + threadIdx.x) >> 5;
  const int nw = (gridDim.x * blockDim.x) >> 5;
  for (int b = gw; b < B_DIM; b += nw) {
    const float* row = cur + (size_t)b * E_DIM;
    float acc = 0.0f;
    for (int j = lane; j < E_DIM; j += 32) acc += row[j] * Wg[j];
#pragma unroll
    for (int off = 16; off; off >>= 1) acc += __shfl_xor(acc, off, 32);
    if (lane == 0) atomicAdd(gatesum, sigmoidf_(acc + bg[0]));
  }
}

// s[b] = [mean(gate)>0.5] * (MOD_SCALE/B) * dot(cur[b], colsum)
__global__ __launch_bounds__(256) void rowmod_kernel(const float* __restrict__ cur,
                                                     const float* __restrict__ colsum,
                                                     const float* __restrict__ gatesum,
                                                     float* __restrict__ svec) {
  const int lane = threadIdx.x & 31;
  const int gw = (blockIdx.x * blockDim.x + threadIdx.x) >> 5;
  const int nw = (gridDim.x * blockDim.x) >> 5;
  const float scale =
      (gatesum[0] * (1.0f / (float)B_DIM) > 0.5f) ? (MOD_SCALE / (float)B_DIM) : 0.0f;
  for (int b = gw; b < B_DIM; b += nw) {
    const float* row = cur + (size_t)b * E_DIM;
    float acc = 0.0f;
    for (int j = lane; j < E_DIM; j += 32) acc += row[j] * colsum[j];
#pragma unroll
    for (int off = 16; off; off >>= 1) acc += __shfl_xor(acc, off, 32);
    if (lane == 0) svec[b] = scale * acc;
  }
}

// ---------------- GEMM1: hid = relu(curh @ W1h.T + b1), f16 out ------------
// M=8192, N=512, K=1024. Block: 8 waves as 4(M) x 2(N); wave tile 16x64.
__global__ __launch_bounds__(256) void mlp1_kernel(const _Float16* __restrict__ curh,
                                                   const _Float16* __restrict__ W1h,
                                                   const float* __restrict__ b1,
                                                   _Float16* __restrict__ hidh) {
  const int lane = threadIdx.x & 31, w = threadIdx.x >> 5;
  const int row0 = blockIdx.x * 64 + (w & 3) * 16;
  const int col0 = blockIdx.y * 128 + (w >> 2) * 64;
  v8f acc[4] = {};
  for (int k0 = 0; k0 < E_DIM; k0 += 32) {
    v16h a = frag_a_f16(curh, E_DIM, row0, k0, lane);
#pragma unroll
    for (int t = 0; t < 4; ++t) {
      v16h b = frag_b_wT_f16(W1h, E_DIM, col0 + t * 16, k0, lane);
      acc[t] = wmma16(a, b, acc[t]);
    }
  }
  const int n = lane & 15, g = lane >> 4;
#pragma unroll
  for (int t = 0; t < 4; ++t) {
    const int col = col0 + t * 16 + n;
    const float bias = b1[col];
#pragma unroll
    for (int v = 0; v < 8; ++v) {
      const int row = row0 + v + g * 8;
      float x = acc[t][v] + bias;
      x = x > 0.0f ? x : 0.0f;
      hidh[(size_t)row * H_DIM + col] = (_Float16)x;
    }
  }
}

// ---------------- GEMM2: sig = tanh(hid @ W2.T + b2) + column sums ---------
// M=8192, N=1024, K=512. sig stored f32; colsum via in-wave 16-row reduction
// (shfl across the lane-16 split of the D layout) + one atomic per column.
__global__ __launch_bounds__(256) void mlp2_kernel(const _Float16* __restrict__ hidh,
                                                   const _Float16* __restrict__ W2h,
                                                   const float* __restrict__ b2,
                                                   float* __restrict__ sig,
                                                   float* __restrict__ colsum) {
  const int lane = threadIdx.x & 31, w = threadIdx.x >> 5;
  const int row0 = blockIdx.x * 64 + (w & 3) * 16;
  const int col0 = blockIdx.y * 128 + (w >> 2) * 64;
  v8f acc[4] = {};
  for (int k0 = 0; k0 < H_DIM; k0 += 32) {
    v16h a = frag_a_f16(hidh, H_DIM, row0, k0, lane);
#pragma unroll
    for (int t = 0; t < 4; ++t) {
      v16h b = frag_b_wT_f16(W2h, H_DIM, col0 + t * 16, k0, lane);
      acc[t] = wmma16(a, b, acc[t]);
    }
  }
  const int n = lane & 15, g = lane >> 4;
#pragma unroll
  for (int t = 0; t < 4; ++t) {
    const int col = col0 + t * 16 + n;
    const float bias = b2[col];
    float csum = 0.0f;
#pragma unroll
    for (int v = 0; v < 8; ++v) {
      const int row = row0 + v + g * 8;
      const float s = tanhf(acc[t][v] + bias);
      sig[(size_t)row * E_DIM + col] = s;
      csum += s;
    }
    csum += __shfl_xor(csum, 16, 32);  // rows 0-7 + rows 8-15 of the tile
    if (g == 0) atomicAdd(&colsum[col], csum);
  }
}

// ---------------- GEMM3 fused GRU: cur_next = (1-z)*n + sig ----------------
// gi[b, e + gE] = (cur @ W_ih.T)[b, e+gE] + b_ih[e+gE] + s[b];  gh = b_hh.
// Block: 8 waves as 4(M) x 2(N); wave tile 16 rows x 32 cols x 3 gates
// (6 f32 accumulators). Writes cur_next directly — gi never hits memory.
__global__ __launch_bounds__(256) void gru_kernel(const _Float16* __restrict__ curh,
                                                  const _Float16* __restrict__ Wihh,
                                                  const float* __restrict__ bih,
                                                  const float* __restrict__ bhh,
                                                  const float* __restrict__ svec,
                                                  const float* __restrict__ sig,
                                                  float* __restrict__ out) {
  const int lane = threadIdx.x & 31, w = threadIdx.x >> 5;
  const int row0 = blockIdx.x * 64 + (w & 3) * 16;
  const int col0 = blockIdx.y * 64 + (w >> 2) * 32;
  v8f aR[2] = {}, aZ[2] = {}, aN[2] = {};
  for (int k0 = 0; k0 < E_DIM; k0 += 32) {
    v16h a = frag_a_f16(curh, E_DIM, row0, k0, lane);
#pragma unroll
    for (int t = 0; t < 2; ++t) {
      const int nb = col0 + t * 16;
      v16h bR = frag_b_wT_f16(Wihh, E_DIM, nb, k0, lane);
      aR[t] = wmma16(a, bR, aR[t]);
      v16h bZ = frag_b_wT_f16(Wihh, E_DIM, E_DIM + nb, k0, lane);
      aZ[t] = wmma16(a, bZ, aZ[t]);
      v16h bN = frag_b_wT_f16(Wihh, E_DIM, 2 * E_DIM + nb, k0, lane);
      aN[t] = wmma16(a, bN, aN[t]);
    }
  }
  const int n = lane & 15, g = lane >> 4;
  float sb[8];
#pragma unroll
  for (int v = 0; v < 8; ++v) sb[v] = svec[row0 + v + g * 8];
#pragma unroll
  for (int t = 0; t < 2; ++t) {
    const int col = col0 + t * 16 + n;
    const float biR = bih[col], biZ = bih[E_DIM + col], biN = bih[2 * E_DIM + col];
    const float hR = bhh[col], hZ = bhh[E_DIM + col], hN = bhh[2 * E_DIM + col];
#pragma unroll
    for (int v = 0; v < 8; ++v) {
      const int row = row0 + v + g * 8;
      const float r = sigmoidf_(aR[t][v] + biR + sb[v] + hR);
      const float z = sigmoidf_(aZ[t][v] + biZ + sb[v] + hZ);
      const float nn = tanhf(aN[t][v] + biN + sb[v] + r * hN);
      out[(size_t)row * E_DIM + col] =
          (1.0f - z) * nn + sig[(size_t)row * E_DIM + col];
    }
  }
}

// ---------------------------------------------------------------------------
static inline unsigned cvt_grid(unsigned long long n) {
  return (unsigned)((n / 8ull + 255ull) / 256ull);
}

extern "C" void kernel_launch(void* const* d_in, const int* in_sizes, int n_in,
                              void* d_out, int out_size, void* d_ws, size_t ws_size,
                              hipStream_t stream) {
  (void)in_sizes; (void)n_in; (void)out_size; (void)ws_size;
  const float* x    = (const float*)d_in[0];
  const float* W_ih = (const float*)d_in[1];
  /* W_hh (d_in[2]) provably unused: h_prev == 0 */
  const float* b_ih = (const float*)d_in[3];
  const float* b_hh = (const float*)d_in[4];
  const float* W1   = (const float*)d_in[5];
  const float* b1   = (const float*)d_in[6];
  const float* W2   = (const float*)d_in[7];
  const float* b2   = (const float*)d_in[8];
  const float* Wg   = (const float*)d_in[9];
  const float* bg   = (const float*)d_in[10];

  // workspace layout
  char* p = (char*)d_ws;
  float* sig = (float*)p;        p += (size_t)B_DIM * E_DIM * sizeof(float);
  float* cur1 = (float*)p;       p += (size_t)B_DIM * E_DIM * sizeof(float);
  float* cur2 = (float*)p;       p += (size_t)B_DIM * E_DIM * sizeof(float);
  _Float16* curh = (_Float16*)p; p += (size_t)B_DIM * E_DIM * sizeof(_Float16);
  _Float16* hidh = (_Float16*)p; p += (size_t)B_DIM * H_DIM * sizeof(_Float16);
  _Float16* Wihh = (_Float16*)p; p += (size_t)L_NUM * 3 * E_DIM * E_DIM * sizeof(_Float16);
  _Float16* W1h  = (_Float16*)p; p += (size_t)L_NUM * H_DIM * E_DIM * sizeof(_Float16);
  _Float16* W2h  = (_Float16*)p; p += (size_t)L_NUM * E_DIM * H_DIM * sizeof(_Float16);
  float* svec = (float*)p;       p += (size_t)B_DIM * sizeof(float);
  float* colsum = (float*)p;     p += (size_t)E_DIM * sizeof(float);
  float* gatesum = (float*)p;    p += 256;

  // one-shot weight conversion (f32 -> f16)
  const unsigned long long nWih = (unsigned long long)L_NUM * 3 * E_DIM * E_DIM;
  const unsigned long long nW1  = (unsigned long long)L_NUM * H_DIM * E_DIM;
  const unsigned long long nW2  = (unsigned long long)L_NUM * E_DIM * H_DIM;
  cvt_f16_kernel<<<cvt_grid(nWih), 256, 0, stream>>>(W_ih, Wihh, nWih);
  cvt_f16_kernel<<<cvt_grid(nW1), 256, 0, stream>>>(W1, W1h, nW1);
  cvt_f16_kernel<<<cvt_grid(nW2), 256, 0, stream>>>(W2, W2h, nW2);

  const unsigned long long nCur = (unsigned long long)B_DIM * E_DIM;

  for (int l = 0; l < L_NUM; ++l) {
    const float* curIn = (l == 0) ? x : ((l == 1) ? cur1 : cur2);
    float* curOut = (l == 0) ? cur1 : ((l == 1) ? cur2 : (float*)d_out);

    const _Float16* Wihh_l = Wihh + (size_t)l * 3 * E_DIM * E_DIM;
    const float* bih_l = b_ih + (size_t)l * 3 * E_DIM;
    const float* bhh_l = b_hh + (size_t)l * 3 * E_DIM;
    const _Float16* W1h_l = W1h + (size_t)l * H_DIM * E_DIM;
    const float* b1_l  = b1 + (size_t)l * H_DIM;
    const _Float16* W2h_l = W2h + (size_t)l * E_DIM * H_DIM;
    const float* b2_l  = b2 + (size_t)l * E_DIM;
    const float* Wg_l  = Wg + (size_t)l * E_DIM;
    const float* bg_l  = bg + (size_t)l;

    zero_kernel<<<dim3((E_DIM + 256) / 256), 256, 0, stream>>>(colsum, gatesum);
    cvt_f16_kernel<<<cvt_grid(nCur), 256, 0, stream>>>(curIn, curh, nCur);
    gate_kernel<<<dim3(256), 256, 0, stream>>>(curIn, Wg_l, bg_l, gatesum);
    mlp1_kernel<<<dim3(B_DIM / 64, H_DIM / 128), 256, 0, stream>>>(curh, W1h_l, b1_l, hidh);
    mlp2_kernel<<<dim3(B_DIM / 64, E_DIM / 128), 256, 0, stream>>>(hidh, W2h_l, b2_l, sig, colsum);
    rowmod_kernel<<<dim3(256), 256, 0, stream>>>(curIn, colsum, gatesum, svec);
    gru_kernel<<<dim3(B_DIM / 64, E_DIM / 64), 256, 0, stream>>>(
        curh, Wihh_l, bih_l, bhh_l, svec, sig, curOut);
  }
}
